// SemanticAssignmentModule_2817498546459
// MI455X (gfx1250) — matrix operand order repaired and checked by reading, not compiled, not run on hardware
//
#include <hip/hip_runtime.h>
#include <hip/hip_bf16.h>

typedef __attribute__((ext_vector_type(16))) __bf16 v16bf;
typedef __attribute__((ext_vector_type(8)))  float  v8f;

#define CF 1984
#define NN 1024
#define BATCH 8

union V16 {
  v16bf v;
  unsigned short u[16];
  uint4 q[2];
};

__device__ __forceinline__ unsigned short f2bf(float f) {
  unsigned int x = __float_as_uint(f);
  unsigned int r = 0x7FFFu + ((x >> 16) & 1u);   // round-to-nearest-even
  return (unsigned short)((x + r) >> 16);
}

// ---------------------------------------------------------------------------
// Pass 1: L2-normalize each feature row (t and r) and emit bf16 copies.
// One block per row; block reduction for sum of squares.
// ---------------------------------------------------------------------------
__global__ __launch_bounds__(256) void norm_bf16_kernel(
    const float* __restrict__ t, const float* __restrict__ r,
    unsigned short* __restrict__ tn, unsigned short* __restrict__ rn) {
  int row = blockIdx.x;                       // 0..16383 (first 8192 = t)
  const float* src;
  unsigned short* dst;
  if (row < BATCH * NN) {
    src = t + (size_t)row * CF;  dst = tn + (size_t)row * CF;
  } else {
    int rr = row - BATCH * NN;
    src = r + (size_t)rr * CF;   dst = rn + (size_t)rr * CF;
  }
  __shared__ float red[256];
  float ss = 0.f;
  for (int i = threadIdx.x; i < CF; i += 256) { float v = src[i]; ss += v * v; }
  red[threadIdx.x] = ss;
  __syncthreads();
  for (int s = 128; s > 0; s >>= 1) {
    if (threadIdx.x < s) red[threadIdx.x] += red[threadIdx.x + s];
    __syncthreads();
  }
  float scale = 1.0f / (sqrtf(red[0]) + 1e-6f);
  for (int i = threadIdx.x; i < CF; i += 256) dst[i] = f2bf(src[i] * scale);
}

// ---------------------------------------------------------------------------
// Pass 2: cost volume C[b,n,m] = tn[b,n,:] . rn[b,m,:]  (bf16 WMMA, f32 acc)
// Block = 128 threads = 4 waves; block tile = 64x64, wave tile = 32x32 (2x2 WMMA).
// A layout (16x32 bf16): lane<16 -> row M=lane, K 0-15; lane>=16 -> K 16-31
// B layout (32x16 bf16): lane<16 -> col N=lane, K 0-15;        lane>=16 -> K 16-31
// ---------------------------------------------------------------------------
__global__ __launch_bounds__(128) void costvol_wmma_kernel(
    const unsigned short* __restrict__ tn, const unsigned short* __restrict__ rn,
    float* __restrict__ Cout) {
  const int b    = blockIdx.z;
  const int wave = threadIdx.x >> 5;
  const int lane = threadIdx.x & 31;
  const int lr   = lane & 15;
  const int hi   = lane >> 4;
  const int nb   = blockIdx.y * 64 + (wave >> 1) * 32;
  const int mb   = blockIdx.x * 64 + (wave & 1) * 32;

  const unsigned short* tbase = tn + (size_t)b * NN * CF;
  const unsigned short* rbase = rn + (size_t)b * NN * CF;

  v8f acc00 = {}, acc01 = {}, acc10 = {}, acc11 = {};

  const unsigned short* pa0 = tbase + (size_t)(nb + lr) * CF      + (hi ? 16 : 0);
  const unsigned short* pa1 = tbase + (size_t)(nb + 16 + lr) * CF + (hi ? 16 : 0);
  const unsigned short* pb0 = rbase + (size_t)(mb + lr) * CF      + (hi ? 16 : 0);
  const unsigned short* pb1 = rbase + (size_t)(mb + 16 + lr) * CF + (hi ? 16 : 0);

  for (int k0 = 0; k0 < CF; k0 += 32) {
    V16 a0, a1, b0, b1;
    a0.q[0] = *(const uint4*)(pa0 + k0);      a0.q[1] = *(const uint4*)(pa0 + k0 + 8);
    a1.q[0] = *(const uint4*)(pa1 + k0);      a1.q[1] = *(const uint4*)(pa1 + k0 + 8);
    b0.q[0] = *(const uint4*)(pb0 + k0);      b0.q[1] = *(const uint4*)(pb0 + k0 + 8);
    b1.q[0] = *(const uint4*)(pb1 + k0);      b1.q[1] = *(const uint4*)(pb1 + k0 + 8);
    acc00 = __builtin_amdgcn_wmma_f32_16x16x32_bf16(false, a0.v, false, b0.v, (short)0, acc00, false, false);
    acc01 = __builtin_amdgcn_wmma_f32_16x16x32_bf16(false, a0.v, false, b1.v, (short)0, acc01, false, false);
    acc10 = __builtin_amdgcn_wmma_f32_16x16x32_bf16(false, a1.v, false, b0.v, (short)0, acc10, false, false);
    acc11 = __builtin_amdgcn_wmma_f32_16x16x32_bf16(false, a1.v, false, b1.v, (short)0, acc11, false, false);
  }

  // D layout: VGPR r -> row M = r + 8*hi; lane lr -> col
  float* cb = Cout + (size_t)b * NN * NN;
#pragma unroll
  for (int r = 0; r < 8; ++r) {
    int row0 = nb + r + hi * 8;
    int row1 = nb + 16 + r + hi * 8;
    cb[(size_t)row0 * NN + mb + lr]      = acc00[r];
    cb[(size_t)row0 * NN + mb + 16 + lr] = acc01[r];
    cb[(size_t)row1 * NN + mb + lr]      = acc10[r];
    cb[(size_t)row1 * NN + mb + 16 + lr] = acc11[r];
  }
}

// ---------------------------------------------------------------------------
// Generic direct 3x3 conv + bias + relu (NHWC / HWIO), used for small convs.
// ---------------------------------------------------------------------------
__global__ __launch_bounds__(256) void conv3x3_relu_kernel(
    const float* __restrict__ x, const float* __restrict__ w,
    const float* __restrict__ bias, float* __restrict__ y,
    int Bn, int IH, int IW, int CI, int OH, int OW, int CO,
    int stride, int padlow) {
  size_t idx = (size_t)blockIdx.x * blockDim.x + threadIdx.x;
  size_t total = (size_t)Bn * OH * OW * CO;
  if (idx >= total) return;
  int co = idx % CO;
  size_t t = idx / CO;
  int ow = t % OW; t /= OW;
  int oh = t % OH;
  int b  = t / OH;
  float acc = bias[co];
  for (int kh = 0; kh < 3; ++kh) {
    int ih = oh * stride + kh - padlow;
    if ((unsigned)ih >= (unsigned)IH) continue;
    for (int kw = 0; kw < 3; ++kw) {
      int iw = ow * stride + kw - padlow;
      if ((unsigned)iw >= (unsigned)IW) continue;
      const float* xp = x + (((size_t)b * IH + ih) * IW + iw) * CI;
      const float* wp = w + (size_t)(kh * 3 + kw) * CI * CO + co;
      for (int ci = 0; ci < CI; ++ci) acc += xp[ci] * wp[(size_t)ci * CO];
    }
  }
  y[idx] = fmaxf(acc, 0.f);
}

// f_rab2 fp32 [b,m,c] -> bf16 channel-major [b,c,m] (B operand for attention)
__global__ __launch_bounds__(256) void frab_transpose_kernel(
    const float* __restrict__ src, unsigned short* __restrict__ dst) {
  size_t idx = (size_t)blockIdx.x * blockDim.x + threadIdx.x; // b*128*1024
  if (idx >= (size_t)BATCH * 128 * NN) return;
  int m = idx % NN;
  int c = (idx / NN) % 128;
  int b = idx / (NN * 128);
  dst[idx] = f2bf(src[((size_t)b * NN + m) * 128 + c]);
}

// ---------------------------------------------------------------------------
// Softmax row stats over C (max, sum of exp) — one block per row.
// ---------------------------------------------------------------------------
__global__ __launch_bounds__(256) void softmax_stats_kernel(
    const float* __restrict__ C, float* __restrict__ rmax, float* __restrict__ rsum) {
  int row = blockIdx.x;                 // b*N + n
  const float* p = C + (size_t)row * NN;
  __shared__ float red[256];
  float m = -3.4e38f;
  for (int i = threadIdx.x; i < NN; i += 256) m = fmaxf(m, p[i]);
  red[threadIdx.x] = m;
  __syncthreads();
  for (int s = 128; s > 0; s >>= 1) {
    if (threadIdx.x < s) red[threadIdx.x] = fmaxf(red[threadIdx.x], red[threadIdx.x + s]);
    __syncthreads();
  }
  m = red[0];
  __syncthreads();
  float sum = 0.f;
  for (int i = threadIdx.x; i < NN; i += 256) sum += __expf(p[i] - m);
  red[threadIdx.x] = sum;
  __syncthreads();
  for (int s = 128; s > 0; s >>= 1) {
    if (threadIdx.x < s) red[threadIdx.x] += red[threadIdx.x + s];
    __syncthreads();
  }
  if (threadIdx.x == 0) { rmax[row] = m; rsum[row] = red[0]; }
}

// ---------------------------------------------------------------------------
// Fused softmax * f_rab GEMM (flash style): f_a[b,n,c] = sum_m softmax(C)[n,m] frab[m,c]
// Block = 128 thr = 4 waves; block covers 16 rows x 128 channels (wave: 16x32).
// A tile (16x32 bf16) is exp((C-max))/sum computed in registers per K-step.
// ---------------------------------------------------------------------------
__global__ __launch_bounds__(128) void attn_wmma_kernel(
    const float* __restrict__ Cmat, const float* __restrict__ rmax,
    const float* __restrict__ rsum, const unsigned short* __restrict__ frabT,
    float* __restrict__ fa) {
  const int b    = blockIdx.y;
  const int n0   = blockIdx.x * 16;
  const int wave = threadIdx.x >> 5;
  const int lane = threadIdx.x & 31;
  const int lr   = lane & 15;
  const int hi   = lane >> 4;
  const int c0   = wave * 32;

  const int rowA = n0 + lr;
  const float rm   = rmax[b * NN + rowA];
  const float rinv = 1.0f / rsum[b * NN + rowA];
  const float* crow = Cmat + ((size_t)b * NN + rowA) * NN;
  const unsigned short* fb0 = frabT + ((size_t)b * 128 + c0 + lr) * NN      + (hi ? 16 : 0);
  const unsigned short* fb1 = frabT + ((size_t)b * 128 + c0 + 16 + lr) * NN + (hi ? 16 : 0);

  v8f acc0 = {}, acc1 = {};
  for (int m0 = 0; m0 < NN; m0 += 32) {
    const int kA = m0 + (hi ? 16 : 0);
    V16 a;
#pragma unroll
    for (int e = 0; e < 8; ++e) {
      a.u[e]     = f2bf(__expf(crow[kA + e]      - rm) * rinv);
      a.u[e + 8] = f2bf(__expf(crow[kA + 8 + e] - rm) * rinv);
    }
    V16 b0, b1;
    b0.q[0] = *(const uint4*)(fb0 + m0);  b0.q[1] = *(const uint4*)(fb0 + m0 + 8);
    b1.q[0] = *(const uint4*)(fb1 + m0);  b1.q[1] = *(const uint4*)(fb1 + m0 + 8);
    acc0 = __builtin_amdgcn_wmma_f32_16x16x32_bf16(false, a.v, false, b0.v, (short)0, acc0, false, false);
    acc1 = __builtin_amdgcn_wmma_f32_16x16x32_bf16(false, a.v, false, b1.v, (short)0, acc1, false, false);
  }

  float* fb = fa + (size_t)b * NN * 128;
#pragma unroll
  for (int r = 0; r < 8; ++r) {
    int row = n0 + r + hi * 8;
    fb[(size_t)row * 128 + c0 + lr]      = acc0[r];
    fb[(size_t)row * 128 + c0 + 16 + lr] = acc1[r];
  }
}

// ---------------------------------------------------------------------------
// Classifier head kernels.
// ---------------------------------------------------------------------------
__global__ __launch_bounds__(256) void maxpool_kernel(
    const float* __restrict__ t, float* __restrict__ mp) {
  int idx = blockIdx.x * 256 + threadIdx.x;   // b*CF + c
  if (idx >= BATCH * CF) return;
  int b = idx / CF, c = idx % CF;
  const float* p = t + (size_t)b * NN * CF + c;
  float m = -3.4e38f;
  for (int n = 0; n < NN; ++n) m = fmaxf(m, p[(size_t)n * CF]);
  mp[idx] = m;
}

__global__ __launch_bounds__(256) void dense1_relu_kernel(
    const float* __restrict__ mp, const float* __restrict__ w,
    const float* __restrict__ bias, float* __restrict__ g1) {
  int idx = blockIdx.x * 256 + threadIdx.x;   // b*512 + j
  if (idx >= BATCH * 512) return;
  int b = idx / 512, j = idx % 512;
  const float* m = mp + (size_t)b * CF;
  float acc = bias[j];
  for (int c = 0; c < CF; ++c) acc += m[c] * w[(size_t)c * 512 + j];
  g1[idx] = fmaxf(acc, 0.f);
}

__global__ __launch_bounds__(256) void dense2_softmax_kernel(
    const float* __restrict__ g1, const float* __restrict__ w,
    const float* __restrict__ bias, float* __restrict__ out) {
  int b = blockIdx.x;
  __shared__ float logits[1000];
  __shared__ float red[256];
  const float* g = g1 + (size_t)b * 512;
  for (int j = threadIdx.x; j < 1000; j += 256) {
    float acc = bias[j];
    for (int c = 0; c < 512; ++c) acc += g[c] * w[(size_t)c * 1000 + j];
    logits[j] = acc;
  }
  __syncthreads();
  float m = -3.4e38f;
  for (int j = threadIdx.x; j < 1000; j += 256) m = fmaxf(m, logits[j]);
  red[threadIdx.x] = m;
  __syncthreads();
  for (int s = 128; s > 0; s >>= 1) {
    if (threadIdx.x < s) red[threadIdx.x] = fmaxf(red[threadIdx.x], red[threadIdx.x + s]);
    __syncthreads();
  }
  m = red[0];
  __syncthreads();
  float sum = 0.f;
  for (int j = threadIdx.x; j < 1000; j += 256) sum += __expf(logits[j] - m);
  red[threadIdx.x] = sum;
  __syncthreads();
  for (int s = 128; s > 0; s >>= 1) {
    if (threadIdx.x < s) red[threadIdx.x] += red[threadIdx.x + s];
    __syncthreads();
  }
  float inv = 1.0f / red[0];
  for (int j = threadIdx.x; j < 1000; j += 256)
    out[(size_t)b * 1000 + j] = __expf(logits[j] - m) * inv;
}

// ---------------------------------------------------------------------------
extern "C" void kernel_launch(void* const* d_in, const int* in_sizes, int n_in,
                              void* d_out, int out_size, void* d_ws, size_t ws_size,
                              hipStream_t stream) {
  const float* t_lum  = (const float*)d_in[0];
  const float* r_lum  = (const float*)d_in[1];
  const float* r_ab   = (const float*)d_in[2];
  const float* w_rab1 = (const float*)d_in[3];
  const float* b_rab1 = (const float*)d_in[4];
  const float* w_rab2 = (const float*)d_in[5];
  const float* b_rab2 = (const float*)d_in[6];
  const float* w_fa1  = (const float*)d_in[7];
  const float* b_fa1  = (const float*)d_in[8];
  const float* w_fa2  = (const float*)d_in[9];
  const float* b_fa2  = (const float*)d_in[10];
  const float* w_fa3  = (const float*)d_in[11];
  const float* b_fa3  = (const float*)d_in[12];
  const float* w_d1   = (const float*)d_in[13];
  const float* b_d1   = (const float*)d_in[14];
  const float* w_d2   = (const float*)d_in[15];
  const float* b_d2   = (const float*)d_in[16];

  // Output tuple layout: (g_tl[8,1000], C[8,1024,1024], f_s1[8,4,4,512],
  //                       f_s2[8,8,8,256], f_s3[8,16,16,128])
  float* out  = (float*)d_out;
  float* g_tl = out;
  float* Cmat = out + 8000;
  float* f_s1 = out + 8396608;
  float* f_s2 = out + 8462144;
  float* f_s3 = out + 8593216;

  // Workspace layout (bytes, 256-aligned)
  char* ws = (char*)d_ws;
  unsigned short* tn_bf  = (unsigned short*)(ws + 0);          // 8*1024*1984*2 = 32,505,856
  unsigned short* rn_bf  = (unsigned short*)(ws + 32505856);   // 32,505,856
  unsigned short* frabT  = (unsigned short*)(ws + 65011712);   // 8*128*1024*2  =  2,097,152
  float*          frab1  = (float*)(ws + 67108864);            // 8*64*64*64*4  =  8,388,608
  float*          frab2f = (float*)(ws + 75497472);            // 8*32*32*128*4 =  4,194,304
  float*          f_a    = (float*)(ws + 79691776);            // 8*1024*128*4  =  4,194,304
  float*          rowmax = (float*)(ws + 83886080);            // 8192*4
  float*          rowsum = (float*)(ws + 83918848);            // 8192*4
  float*          mp     = (float*)(ws + 83951616);            // 8*1984*4
  float*          g1     = (float*)(ws + 84015104);            // 8*512*4

  // 1) normalize + bf16 convert (t and r rows)
  norm_bf16_kernel<<<2 * BATCH * NN, 256, 0, stream>>>(t_lum, r_lum, tn_bf, rn_bf);

  // 2) cost volume via bf16 WMMA -> C output
  costvol_wmma_kernel<<<dim3(16, 16, BATCH), 128, 0, stream>>>(tn_bf, rn_bf, Cmat);

  // 3) chroma branch conv1: [8,64,64,2] -> [8,64,64,64], s1 SAME (pad 1)
  conv3x3_relu_kernel<<<(BATCH * 64 * 64 * 64 + 255) / 256, 256, 0, stream>>>(
      r_ab, w_rab1, b_rab1, frab1, BATCH, 64, 64, 2, 64, 64, 64, 1, 1);

  // 4) chroma branch conv2: [8,64,64,64] -> [8,32,32,128], s2 SAME (pad_low 0)
  conv3x3_relu_kernel<<<(BATCH * 32 * 32 * 128 + 255) / 256, 256, 0, stream>>>(
      frab1, w_rab2, b_rab2, frab2f, BATCH, 64, 64, 64, 32, 32, 128, 2, 0);

  // 5) transpose f_rab to channel-major bf16 for the attention B operand
  frab_transpose_kernel<<<(BATCH * 128 * NN + 255) / 256, 256, 0, stream>>>(frab2f, frabT);

  // 6) softmax row stats of C
  softmax_stats_kernel<<<BATCH * NN, 256, 0, stream>>>(Cmat, rowmax, rowsum);

  // 7) fused softmax x f_rab attention GEMM (bf16 WMMA) -> f_a
  attn_wmma_kernel<<<dim3(NN / 16, BATCH), 128, 0, stream>>>(Cmat, rowmax, rowsum, frabT, f_a);

  // 8) pyramid convs (all s2 SAME, pad_low 0)
  conv3x3_relu_kernel<<<(BATCH * 16 * 16 * 128 + 255) / 256, 256, 0, stream>>>(
      f_a, w_fa1, b_fa1, f_s3, BATCH, 32, 32, 128, 16, 16, 128, 2, 0);
  conv3x3_relu_kernel<<<(BATCH * 8 * 8 * 256 + 255) / 256, 256, 0, stream>>>(
      f_s3, w_fa2, b_fa2, f_s2, BATCH, 16, 16, 128, 8, 8, 256, 2, 0);
  conv3x3_relu_kernel<<<(BATCH * 4 * 4 * 512 + 255) / 256, 256, 0, stream>>>(
      f_s2, w_fa3, b_fa3, f_s1, BATCH, 8, 8, 256, 4, 4, 512, 2, 0);

  // 9) classifier head
  maxpool_kernel<<<(BATCH * CF + 255) / 256, 256, 0, stream>>>(t_lum, mp);
  dense1_relu_kernel<<<(BATCH * 512 + 255) / 256, 256, 0, stream>>>(mp, w_d1, b_d1, g1);
  dense2_softmax_kernel<<<BATCH, 256, 0, stream>>>(g1, w_d2, b_d2, g_tl);
}
